// SequencePulse_64424509440472
// MI455X (gfx1250) — matrix operand order repaired and checked by reading, not compiled, not run on hardware
//
#include <hip/hip_runtime.h>

typedef __attribute__((ext_vector_type(16))) _Float16 v16h;
typedef __attribute__((ext_vector_type(8)))  _Float16 v8h;
typedef __attribute__((ext_vector_type(8)))  float    v8f;

#define SH  264  // row stride (halves) for 16x256 f16 LDS matrices (=528B, 16B aligned, conflict-friendly)
#define STH 24   // row stride (halves) for transposed 256x16 and 16x16 attn buffers (=48B)

__device__ __forceinline__ v16h cat8(v8h lo, v8h hi) {
  return __builtin_shufflevector(lo, hi, 0, 1, 2, 3, 4, 5, 6, 7, 8, 9, 10, 11,
                                 12, 13, 14, 15);
}
__device__ __forceinline__ v8f wmma16(v16h a, v16h b, v8f c) {
  return __builtin_amdgcn_wmma_f32_16x16x32_f16(false, a, false, b, (short)0, c,
                                                false, false);
}
// A-fragment (this code's contiguous k order, shared with B): halves 0-15 <- k=kbase+16g+h.
// Two contiguous 16B chunks.
__device__ __forceinline__ v16h loadA(const _Float16* row, int kbase, int g) {
  const v8h* p0 = (const v8h*)(row + kbase + 16 * g);
  const v8h* p1 = (const v8h*)(row + kbase + 16 * g + 8);
  return cat8(*p0, *p1);
}
// B-fragment (this code's contiguous k order, shared with A): halves 0-15 <- k=kbase+16g+h.
// One contiguous 32B run (row = column of B, i.e. row of B^T source).
__device__ __forceinline__ v16h loadB(const _Float16* row, int kbase, int g) {
  const v8h* p = (const v8h*)(row + kbase + 16 * g);
  return cat8(p[0], p[1]);
}

// ---------------------------------------------------------------------------
// Pre-pass 1: f32 weight [K,N] -> f16 tiles in WMMA B-fragment order.
// Tile linear index = nt*KT + kt ; within tile off = lane*16 + h,
// lane = 16*(kl>>4) + nl, h = kl&15  (contiguous k order: lanes0-15 K0-15, lanes16-31 K16-31)
// ---------------------------------------------------------------------------
__global__ void convert_tiled(const float* __restrict__ src,
                              _Float16* __restrict__ dst, int K, int N) {
  int KT = K >> 5;
  int total = K * N;
  for (int idx = blockIdx.x * blockDim.x + threadIdx.x; idx < total;
       idx += gridDim.x * blockDim.x) {
    int k = idx / N, n = idx - k * N;
    int kt = k >> 5, kl = k & 31, nt = n >> 4, nl = n & 15;
    int lane = ((kl >> 4) << 4) + nl;
    int h = kl & 15;
    size_t off = ((size_t)(nt * KT + kt) << 9) + lane * 16 + h;
    dst[off] = (_Float16)src[idx];
  }
}

// ---------------------------------------------------------------------------
// Pre-pass 2: inp_all = x @ W_in + b_in  ([16384,512]@[512,256]), one-time.
// ---------------------------------------------------------------------------
__global__ void inp_prepass(const float* __restrict__ x,
                            const float* __restrict__ W_in,
                            const float* __restrict__ b_in,
                            float* __restrict__ inp_all) {
  const int total = 16384 * 256;
  for (int idx = blockIdx.x * blockDim.x + threadIdx.x; idx < total;
       idx += gridDim.x * blockDim.x) {
    int r = idx >> 8, c = idx & 255;
    const float* xr = x + (size_t)r * 512;
    float acc = b_in[c];
#pragma unroll 4
    for (int k = 0; k < 512; ++k) acc += xr[k] * W_in[k * 256 + c];
    inp_all[idx] = acc;
  }
}

// ---------------------------------------------------------------------------
// Persistent scan: one workgroup (16 wave32s) per batch element. Wave w owns
// output tile nt=w in every phase, so its state/z/nw tiles stay in registers.
// ---------------------------------------------------------------------------
__global__ __launch_bounds__(512) void gru_scan(
    const float* __restrict__ x, const float* __restrict__ W_in,
    const float* __restrict__ b_in, const float* __restrict__ b_z,
    const float* __restrict__ b_r, const float* __restrict__ b_h,
    const float* __restrict__ b_c, const float* __restrict__ W_out,
    const float* __restrict__ b_out, const float* __restrict__ init_states,
    const _Float16* __restrict__ Wz_t, const _Float16* __restrict__ Wr_t,
    const _Float16* __restrict__ Wh_t, const _Float16* __restrict__ Wc_t,
    const float* __restrict__ inp_all, int use_pre, float* __restrict__ out) {
  const int b = blockIdx.x;
  const int tid = threadIdx.x;
  const int lane = tid & 31;
  const int wave = tid >> 5;
  const int g = lane >> 4;
  const int ln16 = lane & 15;
  const int col = (wave << 4) + ln16;  // this wave's output column

  __shared__ __align__(16) _Float16 st16[16 * SH];   // states, row-major f16
  __shared__ __align__(16) _Float16 rg16[16 * SH];   // reset gate
  __shared__ __align__(16) _Float16 nw16[16 * SH];   // post-GRU state
  __shared__ __align__(16) _Float16 cm16[16 * SH];   // attention mix
  __shared__ __align__(16) _Float16 nwT16[256 * STH];// nw transposed [d][n]
  __shared__ __align__(16) _Float16 at16[16 * STH];  // softmaxed attention
  __shared__ __align__(16) _Float16 inp16[272];      // input projection (f16)
  __shared__ float sc[16 * 17];                      // raw scores
  __shared__ float xs[512];                          // GEMV fallback buffers
  __shared__ float pin[512];

  // init: states broadcast, both LDS f16 copy and per-wave register tile
  for (int i = tid; i < 4096; i += 512)
    st16[(i >> 8) * SH + (i & 255)] = (_Float16)init_states[i];
  v8f st_reg, z_reg, nw_reg;
#pragma unroll
  for (int v = 0; v < 8; ++v)
    st_reg[v] = init_states[(v + 8 * g) * 256 + col];
  __syncthreads();

  const _Float16* Wz_w = Wz_t + ((size_t)wave << 13) + lane * 16;  // wave*16*512
  const _Float16* Wr_w = Wr_t + ((size_t)wave << 13) + lane * 16;
  const _Float16* Wh_w = Wh_t + ((size_t)wave << 13) + lane * 16;
  const _Float16* Wc_w = Wc_t + ((size_t)wave << 12) + lane * 16;  // wave*8*512

  for (int t = 0; t < 512; ++t) {
    // ---- phase 1: inp (precomputed) or in-kernel GEMV fallback -> inp16
    if (use_pre) {
      if (tid < 256)
        inp16[tid] = (_Float16)inp_all[(((size_t)b << 9) + t) * 256 + tid];
    } else {
      xs[tid] = x[(((size_t)b << 9) + t) * 512 + tid];
      __syncthreads();
      {
        int c = tid & 255, p = tid >> 8;
        int k0 = p << 8;
        float acc = 0.f;
#pragma unroll 4
        for (int k = 0; k < 256; ++k)
          acc += xs[k0 + k] * W_in[(k0 + k) * 256 + c];
        pin[(p << 8) + c] = acc;
      }
      __syncthreads();
      if (tid < 256)
        inp16[tid] = (_Float16)(pin[tid] + pin[256 + tid] + b_in[tid]);
    }
    __syncthreads();

    // ---- phase 2: r and z gates share one A-fragment stream, two WMMA chains
    {
      v8f racc = {}, zacc = {};
#pragma unroll 2
      for (int kt = 0; kt < 16; ++kt) {
        v16h a = (kt < 8) ? loadA(st16 + ln16 * SH, kt << 5, g)
                          : loadA(inp16, (kt - 8) << 5, g);
        racc = wmma16(a, *(const v16h*)(Wr_w + (kt << 9)), racc);
        zacc = wmma16(a, *(const v16h*)(Wz_w + (kt << 9)), zacc);
      }
      float rb = b_r[col];
#pragma unroll
      for (int v = 0; v < 8; ++v)
        rg16[(v + 8 * g) * SH + col] =
            (_Float16)(1.f / (1.f + __expf(-(racc[v] + rb))));
      float zb = b_z[col];
#pragma unroll
      for (int v = 0; v < 8; ++v)
        z_reg[v] = 1.f / (1.f + __expf(-(zacc[v] + zb)));
    }
    __syncthreads();

    // ---- phase 3: h = tanh([r*st|inp]@W_h+b_h); GRU update (regs + LDS f16)
    {
      v8f hacc = {};
#pragma unroll 2
      for (int kt = 0; kt < 16; ++kt) {
        v16h a;
        if (kt < 8) {
          v16h ra = loadA(rg16 + ln16 * SH, kt << 5, g);
          v16h sa = loadA(st16 + ln16 * SH, kt << 5, g);
          a = ra * sa;  // packed f16 multiply
        } else {
          a = loadA(inp16, (kt - 8) << 5, g);
        }
        hacc = wmma16(a, *(const v16h*)(Wh_w + (kt << 9)), hacc);
      }
      float hb = b_h[col];
      v8h pk;
#pragma unroll
      for (int v = 0; v < 8; ++v) {
        float e2 = __expf(2.f * (hacc[v] + hb));
        float hc = (e2 - 1.f) / (e2 + 1.f);
        float nv = (1.f - z_reg[v]) * st_reg[v] + z_reg[v] * hc;
        nw_reg[v] = nv;
        nw16[(v + 8 * g) * SH + col] = (_Float16)nv;
        pk[v] = (_Float16)nv;
      }
      *(v8h*)(nwT16 + col * STH + 8 * g) = pk;  // packed transposed store
    }
    __syncthreads();

    // ---- phase 4: scores = nw@nw^T/sqrt(D); softmax (wave 0 only)
    if (wave == 0) {
      v8f acc = {};
#pragma unroll
      for (int kt = 0; kt < 8; ++kt) {
        v16h a = loadA(nw16 + ln16 * SH, kt << 5, g);
        v16h bb = loadB(nw16 + ln16 * SH, kt << 5, g);  // B[k,m]=nw[m,k]
        acc = wmma16(a, bb, acc);
      }
#pragma unroll
      for (int v = 0; v < 8; ++v)
        sc[(v + 8 * g) * 17 + ln16] = acc[v] * 0.0625f;
      asm volatile("s_wait_dscnt 0x0" ::: "memory");  // same-wave DS ordering
      if (lane < 16) {
        int r = lane;
        float mx = sc[r * 17];
#pragma unroll
        for (int c = 1; c < 16; ++c) mx = fmaxf(mx, sc[r * 17 + c]);
        float sum = 0.f;
#pragma unroll
        for (int c = 0; c < 16; ++c) sum += __expf(sc[r * 17 + c] - mx);
        float inv = 1.f / sum;
#pragma unroll
        for (int c = 0; c < 16; ++c)
          at16[r * STH + c] = (_Float16)(__expf(sc[r * 17 + c] - mx) * inv);
      }
    }
    __syncthreads();

    // ---- phase 5: comm = attn @ nw  (K=16 zero-padded to 32)
    {
      v8h zz = {};
      // A: rows of attn; k = h+16g, g==1 -> k>=16 -> zero
      v16h a = (g == 0) ? cat8(*(const v8h*)(at16 + ln16 * STH),
                               *(const v8h*)(at16 + ln16 * STH + 8))
                        : cat8(zz, zz);
      // B: nw[k][col] = nwT16[col*STH + k], contiguous; g==1 -> k>=16 -> zero
      v16h bfull = cat8(*(const v8h*)(nwT16 + col * STH),
                        *(const v8h*)(nwT16 + col * STH + 8));
      v16h bb = (g == 0) ? bfull : cat8(zz, zz);
      v8f acc = {};
      acc = wmma16(a, bb, acc);
#pragma unroll
      for (int v = 0; v < 8; ++v)
        cm16[(v + 8 * g) * SH + col] = (_Float16)acc[v];
    }
    __syncthreads();

    // ---- phase 6: st = nw + comm @ W_c + b_c  (regs + LDS f16)
    {
      v8f acc = {};
#pragma unroll
      for (int kt = 0; kt < 8; ++kt) {
        v16h a = loadA(cm16 + ln16 * SH, kt << 5, g);
        acc = wmma16(a, *(const v16h*)(Wc_w + (kt << 9)), acc);
      }
      float cb = b_c[col];
#pragma unroll
      for (int v = 0; v < 8; ++v) {
        float sv = nw_reg[v] + acc[v] + cb;
        st_reg[v] = sv;
        st16[(v + 8 * g) * SH + col] = (_Float16)sv;
      }
    }
    __syncthreads();
  }

  // ---- final (once): out[b,:] = st.flatten() @ W_out + b_out
  {
    int oc = tid;
    float acc = b_out[oc];
#pragma unroll 4
    for (int k = 0; k < 4096; ++k)
      acc += (float)st16[(k >> 8) * SH + (k & 255)] * W_out[(size_t)k * 512 + oc];
    out[((size_t)b << 9) + oc] = acc;
  }
}

extern "C" void kernel_launch(void* const* d_in, const int* in_sizes, int n_in,
                              void* d_out, int out_size, void* d_ws,
                              size_t ws_size, hipStream_t stream) {
  (void)in_sizes; (void)n_in; (void)out_size;
  const float* x           = (const float*)d_in[0];
  const float* W_in        = (const float*)d_in[1];
  const float* b_in        = (const float*)d_in[2];
  const float* W_z         = (const float*)d_in[3];
  const float* b_z         = (const float*)d_in[4];
  const float* W_r         = (const float*)d_in[5];
  const float* b_r         = (const float*)d_in[6];
  const float* W_h         = (const float*)d_in[7];
  const float* b_h         = (const float*)d_in[8];
  const float* W_c         = (const float*)d_in[9];
  const float* b_c         = (const float*)d_in[10];
  const float* W_out       = (const float*)d_in[11];
  const float* b_out       = (const float*)d_in[12];
  const float* init_states = (const float*)d_in[13];
  float* out = (float*)d_out;

  _Float16* wsh  = (_Float16*)d_ws;
  _Float16* Wz_t = wsh;            // 512*256
  _Float16* Wr_t = wsh + 131072;   // 512*256
  _Float16* Wh_t = wsh + 262144;   // 512*256
  _Float16* Wc_t = wsh + 393216;   // 256*256  (ends at 917504 B)
  float* inp_all = (float*)((char*)d_ws + (1 << 20));
  size_t need_pre = (size_t)(1 << 20) + (size_t)32 * 512 * 256 * sizeof(float);
  int use_pre = (ws_size >= need_pre) ? 1 : 0;

  convert_tiled<<<256, 256, 0, stream>>>(W_z, Wz_t, 512, 256);
  convert_tiled<<<256, 256, 0, stream>>>(W_r, Wr_t, 512, 256);
  convert_tiled<<<256, 256, 0, stream>>>(W_h, Wh_t, 512, 256);
  convert_tiled<<<128, 256, 0, stream>>>(W_c, Wc_t, 256, 256);
  if (use_pre) inp_prepass<<<8192, 256, 0, stream>>>(x, W_in, b_in, inp_all);

  gru_scan<<<dim3(32), dim3(512), 0, stream>>>(
      x, W_in, b_in, b_z, b_r, b_h, b_c, W_out, b_out, init_states, Wz_t, Wr_t,
      Wh_t, Wc_t, inp_all, use_pre, out);
}
